// EGNNLayer_46334107189561
// MI455X (gfx1250) — matrix operand order repaired and checked by reading, not compiled, not run on hardware
//
#include <hip/hip_runtime.h>

#define H    128
#define NIV  32
#define K1   (2*H + NIV)   // 288
#define LDW  136           // padded row stride in bf16 elems (keeps 16B alignment)
#define WAVES 8
#define BLK  (WAVES*32)

typedef __attribute__((ext_vector_type(16))) __bf16 v16bf;
typedef __attribute__((ext_vector_type(8)))  float  v8f;
typedef int v4i_g __attribute__((vector_size(16)));   // matches async builtin param type

union BF16x16 { v16bf v; __bf16 b[16]; unsigned short u[16]; uint4 q[2]; };

#if __has_builtin(__builtin_amdgcn_global_load_async_to_lds_b128)
#define ASYNC_LDS 1
#endif

#if __has_builtin(__builtin_amdgcn_sched_barrier)
#define SCHED_FENCE() __builtin_amdgcn_sched_barrier(0)
#else
#define SCHED_FENCE()
#endif

__device__ __forceinline__ unsigned short f2bf(float f) {
    union { __bf16 b; unsigned short u; } cv;
    cv.b = (__bf16)f;                    // hardware v_cvt bf16 (RNE)
    return cv.u;
}
__device__ __forceinline__ float silu_f(float v) { return v / (1.0f + __expf(-v)); }
__device__ __forceinline__ float sigm_f(float v) { return 1.0f / (1.0f + __expf(-v)); }

__device__ __forceinline__ v8f vzero8() {
    v8f z;
#pragma unroll
    for (int i = 0; i < 8; ++i) z[i] = 0.0f;
    return z;
}

// B fragment (16-bit, 32x16): lane l holds column N=n0+l%16, element s = K=k0+8*(l/16)+(s&7)+16*(s>>3).
__device__ __forceinline__ v16bf ldsB(const unsigned short* W, int k0, int n0, int lane) {
    const unsigned short* p = W + n0 + (lane & 15);
    const int hb = 8 * (lane >> 4);
    BF16x16 r;
#pragma unroll
    for (int s = 0; s < 16; ++s)
        r.u[s] = p[(unsigned)(k0 + hb + (s & 7) + 16 * (s >> 3)) * LDW];
    return r.v;
}

__device__ __forceinline__ void loadB8(v16bf bf[8], const unsigned short* W, int k0,
                                       int lane) {
#pragma unroll
    for (int t = 0; t < 8; ++t) bf[t] = ldsB(W, k0, t * 16, lane);
}

// A fragment (16-bit, 16x32) from LDS row [M][K]; row baked into rowp.
__device__ __forceinline__ v16bf ldsA(const unsigned short* rowp, int k0, int hl) {
    BF16x16 r;
    r.q[0] = *(const uint4*)(rowp + k0 + 8 * hl);
    r.q[1] = *(const uint4*)(rowp + k0 + 16 + 8 * hl);
    return r.v;
}

// Bulk copy of pre-converted bf16 weights (padded [K][LDW], valid cols 0..127)
// from workspace to LDS, 16B per transfer (async DMA path when available).
__device__ __forceinline__ void copyW(const unsigned short* __restrict__ ws,
                                      unsigned short* __restrict__ lds,
                                      int K, int tid) {
    for (int i = tid; i < K * 16; i += BLK) {
        int r = i >> 4, s = (i & 15) * 8;
#ifdef ASYNC_LDS
        __builtin_amdgcn_global_load_async_to_lds_b128(
            (v4i_g*)(ws + r * LDW + s),
            (v4i_g*)(lds + r * LDW + s),
            0, 0);
#else
        *(uint4*)(lds + r * LDW + s) = *(const uint4*)(ws + r * LDW + s);
#endif
    }
}

__device__ __forceinline__ void waitW() {
#ifdef ASYNC_LDS
    asm volatile("s_wait_asynccnt 0x0" ::: "memory");
#endif
    __syncthreads();
}

// ---------------------------------------------------------------------------
// Prep: convert f32 weights [K][128] -> padded bf16 [K][LDW] in workspace.
// ---------------------------------------------------------------------------
__global__ void egnn_cvt_w(const float* __restrict__ W, unsigned short* __restrict__ S,
                           int K) {
    int i = (blockIdx.x * blockDim.x + threadIdx.x) * 4;
    if (i < K * H) {
        float4 w = *(const float4*)(W + i);
        int r = i >> 7, c = i & 127;
        unsigned long long pk =
            (unsigned long long)f2bf(w.x) |
            ((unsigned long long)f2bf(w.y) << 16) |
            ((unsigned long long)f2bf(w.z) << 32) |
            ((unsigned long long)f2bf(w.w) << 48);
        *(unsigned long long*)(S + r * LDW + c) = pk;
    }
}

__global__ void egnn_zero_f4(float4* __restrict__ p, long long n4) {
    long long i = (long long)blockIdx.x * blockDim.x + threadIdx.x;
    if (i < n4) p[i] = make_float4(0.f, 0.f, 0.f, 0.f);
}

// ---------------------------------------------------------------------------
// Edge kernel: msg = silu(silu([x_s|x_r|inv]@Wm1+bm1)@Wm2+bm2);
// gate = sigmoid(msg@We+be); atomic scatter of msg*gate into aggr[rec].
// Block = 8 waves, 128 edges; wave = 16 edges x 128 cols (8 WMMA C tiles).
// ---------------------------------------------------------------------------
__global__ void __launch_bounds__(BLK) __attribute__((amdgpu_waves_per_eu(2, 2)))
egnn_edge_kernel(const float* __restrict__ x,
                 const int*   __restrict__ adj,
                 const float* __restrict__ inv,
                 const unsigned short* __restrict__ wsW1,
                 const unsigned short* __restrict__ wsW2,
                 const float* __restrict__ bm1, const float* __restrict__ bm2,
                 const float* __restrict__ We,  const float* __restrict__ be,
                 float* __restrict__ aggr, int E)
{
    extern __shared__ unsigned char smem[];
    unsigned short* sW1 = (unsigned short*)smem;            // [288][LDW]
    unsigned short* sW2 = sW1 + K1 * LDW;                   // [128][LDW]
    unsigned char*  pw  = (unsigned char*)(sW2 + H * LDW);  // per-wave union region

    const int tid  = threadIdx.x;
    const int lane = tid & 31;
    const int wv   = tid >> 5;
    const int hl   = lane >> 4;
    const int row  = lane & 15;

    copyW(wsW1, sW1, K1, tid);
    copyW(wsW2, sW2, H, tid);
    waitW();

    unsigned short* sH = (unsigned short*)(pw + wv * (16 * LDW * 4)); // bf16 [16][LDW]
    float*          sM = (float*)(pw + wv * (16 * LDW * 4));          // f32  [16][LDW]

    const int ebase = blockIdx.x * (WAVES * 16) + wv * 16;
    int myEdge = ebase + row;
    int eC     = myEdge < E ? myEdge : E - 1;
    const float* xs = x   + (size_t)adj[eC]     * H;
    const float* xr = x   + (size_t)adj[E + eC] * H;
    const float* iv = inv + (size_t)eC * NIV;

    v8f acc[8];
#pragma unroll
    for (int t = 0; t < 8; ++t) acc[t] = vzero8();

    // ---- GEMM1: K = 288 (9 chunks of 32), software-pipelined B loads ----
    v16bf bp[8];
    loadB8(bp, sW1, 0, lane);
#pragma unroll
    for (int c = 0; c < 9; ++c) {
        v16bf bn[8];
        if (c < 8) loadB8(bn, sW1, (c + 1) * 32, lane);   // prefetch next chunk

        const float* src = (c < 4) ? xs : (c < 8) ? xr : iv;
        int kb = (c < 4) ? c * 32 : (c < 8) ? c * 32 - 128 : 0;
        BF16x16 a;
#pragma unroll
        for (int v2 = 0; v2 < 8; ++v2) {
            int k = kb + 16 * (v2 >> 2) + 8 * hl + 2 * (v2 & 3);
            float2 f = *(const float2*)(src + k);
            a.b[2 * v2]     = (__bf16)f.x;
            a.b[2 * v2 + 1] = (__bf16)f.y;
        }
        SCHED_FENCE();   // keep the prefetch loads above the WMMA block
#pragma unroll
        for (int t = 0; t < 8; ++t) {
            acc[t] = __builtin_amdgcn_wmma_f32_16x16x32_bf16(
                false, a.v, false, bp[t], (short)0, acc[t], false, false);
        }
        if (c < 8) {
#pragma unroll
            for (int t = 0; t < 8; ++t) bp[t] = bn[t];    // SSA rotate (free)
        }
    }

    // bias + SiLU, C layout -> bf16 LDS [M][K]
#pragma unroll
    for (int t = 0; t < 8; ++t) {
        int n = t * 16 + row;
        float bias = bm1[n];
#pragma unroll
        for (int r = 0; r < 8; ++r) {
            int m = r + 8 * hl;
            sH[m * LDW + n] = f2bf(silu_f(acc[t][r] + bias));
        }
        acc[t] = vzero8();
    }
    __builtin_amdgcn_wave_barrier();
    asm volatile("s_wait_dscnt 0x0" ::: "memory");

    // ---- GEMM2: K = 128 (4 chunks), A from LDS, pipelined B loads ----
    loadB8(bp, sW2, 0, lane);
#pragma unroll
    for (int c = 0; c < 4; ++c) {
        v16bf bn[8];
        if (c < 3) loadB8(bn, sW2, (c + 1) * 32, lane);
        v16bf a = ldsA(sH + row * LDW, c * 32, hl);
        SCHED_FENCE();
#pragma unroll
        for (int t = 0; t < 8; ++t) {
            acc[t] = __builtin_amdgcn_wmma_f32_16x16x32_bf16(
                false, a, false, bp[t], (short)0, acc[t], false, false);
        }
        if (c < 3) {
#pragma unroll
            for (int t = 0; t < 8; ++t) bp[t] = bn[t];
        }
    }

    // bias + SiLU -> msg (f32) into LDS (reuses sH region; DS ops in-order per wave)
#pragma unroll
    for (int t = 0; t < 8; ++t) {
        int n = t * 16 + row;
        float bias = bm2[n];
#pragma unroll
        for (int r = 0; r < 8; ++r) {
            int m = r + 8 * hl;
            sM[m * LDW + n] = silu_f(acc[t][r] + bias);
        }
    }
    __builtin_amdgcn_wave_barrier();
    asm volatile("s_wait_dscnt 0x0" ::: "memory");

    // ---- gate + scatter: each lane owns 4 columns ----
    int c0 = lane * 4;
    float4 w4 = *(const float4*)(We + c0);
    float b0 = be[0];
    for (int e = 0; e < 16; ++e) {
        float4 m4 = *(const float4*)(sM + e * LDW + c0);
        float part = m4.x * w4.x + m4.y * w4.y + m4.z * w4.z + m4.w * w4.w;
#pragma unroll
        for (int off = 16; off > 0; off >>= 1)
            part += __shfl_xor(part, off, 32);
        int edge = ebase + e;
        if (edge < E) {
            float g = sigm_f(part + b0);
            float* dst = aggr + (size_t)adj[E + edge] * H + c0;
            unsafeAtomicAdd(dst + 0, m4.x * g);
            unsafeAtomicAdd(dst + 1, m4.y * g);
            unsafeAtomicAdd(dst + 2, m4.z * g);
            unsafeAtomicAdd(dst + 3, m4.w * g);
        }
    }
}

// ---------------------------------------------------------------------------
// Update kernel: out = x + silu([x|aggr]@Wu1+bu1)@Wu2+bu2.
// `out` holds aggr on entry; each wave reads only its own 16 rows before
// overwriting them.
// ---------------------------------------------------------------------------
__global__ void __launch_bounds__(BLK) __attribute__((amdgpu_waves_per_eu(2, 2)))
egnn_update_kernel(const float* __restrict__ x,
                   const unsigned short* __restrict__ wsU1,
                   const unsigned short* __restrict__ wsU2,
                   const float* __restrict__ bu1, const float* __restrict__ bu2,
                   float* __restrict__ out, int N)
{
    extern __shared__ unsigned char smem[];
    unsigned short* sW1 = (unsigned short*)smem;      // [256][LDW]
    unsigned short* sW2 = sW1 + 2 * H * LDW;          // [128][LDW]
    unsigned short* sHa = sW2 + H * LDW;              // 8 x [16][LDW]

    const int tid  = threadIdx.x;
    const int lane = tid & 31;
    const int wv   = tid >> 5;
    const int hl   = lane >> 4;
    const int row  = lane & 15;

    copyW(wsU1, sW1, 2 * H, tid);
    copyW(wsU2, sW2, H, tid);
    waitW();

    unsigned short* sH = sHa + wv * 16 * LDW;

    const int nbase = blockIdx.x * (WAVES * 16) + wv * 16;
    int myNode = nbase + row;
    int nC     = myNode < N ? myNode : N - 1;
    const float* xp = x   + (size_t)nC * H;
    const float* ap = out + (size_t)nC * H;   // aggregated messages

    v8f acc[8];
#pragma unroll
    for (int t = 0; t < 8; ++t) acc[t] = vzero8();

    // ---- GEMM1: K = 256 over [x | aggr], pipelined B loads ----
    v16bf bp[8];
    loadB8(bp, sW1, 0, lane);
#pragma unroll
    for (int c = 0; c < 8; ++c) {
        v16bf bn[8];
        if (c < 7) loadB8(bn, sW1, (c + 1) * 32, lane);

        const float* src = (c < 4) ? xp : ap;
        int kb = (c < 4) ? c * 32 : c * 32 - 128;
        BF16x16 a;
#pragma unroll
        for (int v2 = 0; v2 < 8; ++v2) {
            int k = kb + 16 * (v2 >> 2) + 8 * hl + 2 * (v2 & 3);
            float2 f = *(const float2*)(src + k);
            a.b[2 * v2]     = (__bf16)f.x;
            a.b[2 * v2 + 1] = (__bf16)f.y;
        }
        SCHED_FENCE();
#pragma unroll
        for (int t = 0; t < 8; ++t) {
            acc[t] = __builtin_amdgcn_wmma_f32_16x16x32_bf16(
                false, a.v, false, bp[t], (short)0, acc[t], false, false);
        }
        if (c < 7) {
#pragma unroll
            for (int t = 0; t < 8; ++t) bp[t] = bn[t];
        }
    }

#pragma unroll
    for (int t = 0; t < 8; ++t) {
        int n = t * 16 + row;
        float bias = bu1[n];
#pragma unroll
        for (int r = 0; r < 8; ++r) {
            int m = r + 8 * hl;
            sH[m * LDW + n] = f2bf(silu_f(acc[t][r] + bias));
        }
        acc[t] = vzero8();
    }
    __builtin_amdgcn_wave_barrier();
    asm volatile("s_wait_dscnt 0x0" ::: "memory");

    // ---- GEMM2: K = 128, pipelined B loads ----
    loadB8(bp, sW2, 0, lane);
#pragma unroll
    for (int c = 0; c < 4; ++c) {
        v16bf bn[8];
        if (c < 3) loadB8(bn, sW2, (c + 1) * 32, lane);
        v16bf a = ldsA(sH + row * LDW, c * 32, hl);
        SCHED_FENCE();
#pragma unroll
        for (int t = 0; t < 8; ++t) {
            acc[t] = __builtin_amdgcn_wmma_f32_16x16x32_bf16(
                false, a, false, bp[t], (short)0, acc[t], false, false);
        }
        if (c < 3) {
#pragma unroll
            for (int t = 0; t < 8; ++t) bp[t] = bn[t];
        }
    }

    // epilogue: residual + bias
#pragma unroll
    for (int t = 0; t < 8; ++t) {
        int n = t * 16 + row;
        float b2 = bu2[n];
#pragma unroll
        for (int r = 0; r < 8; ++r) {
            int m = r + 8 * hl;
            int node = nbase + m;
            if (node < N) {
                size_t idx = (size_t)node * H + n;
                out[idx] = x[idx] + acc[t][r] + b2;
            }
        }
    }
}

extern "C" void kernel_launch(void* const* d_in, const int* in_sizes, int n_in,
                              void* d_out, int out_size, void* d_ws, size_t ws_size,
                              hipStream_t stream) {
    const float* x   = (const float*)d_in[0];
    const int*   adj = (const int*)d_in[1];
    const float* inv = (const float*)d_in[2];
    const float* Wm1 = (const float*)d_in[3];
    const float* bm1 = (const float*)d_in[4];
    const float* Wm2 = (const float*)d_in[5];
    const float* bm2 = (const float*)d_in[6];
    const float* We  = (const float*)d_in[7];
    const float* be  = (const float*)d_in[8];
    const float* Wu1 = (const float*)d_in[9];
    const float* bu1 = (const float*)d_in[10];
    const float* Wu2 = (const float*)d_in[11];
    const float* bu2 = (const float*)d_in[12];
    float* out = (float*)d_out;

    const int N = in_sizes[0] / H;
    const int E = in_sizes[1] / 2;

    // workspace layout: padded bf16 weight images
    unsigned short* wsW1 = (unsigned short*)d_ws;            // [288][LDW]
    unsigned short* wsW2 = wsW1 + K1 * LDW;                  // [128][LDW]
    unsigned short* wsU1 = wsW2 + H * LDW;                   // [256][LDW]
    unsigned short* wsU2 = wsU1 + 2 * H * LDW;               // [128][LDW]

    // 0) convert weights to padded bf16 (once per call)
    egnn_cvt_w<<<(K1 * H / 4 + 255) / 256, 256, 0, stream>>>(Wm1, wsW1, K1);
    egnn_cvt_w<<<(H  * H / 4 + 255) / 256, 256, 0, stream>>>(Wm2, wsW2, H);
    egnn_cvt_w<<<(2*H* H / 4 + 255) / 256, 256, 0, stream>>>(Wu1, wsU1, 2 * H);
    egnn_cvt_w<<<(H  * H / 4 + 255) / 256, 256, 0, stream>>>(Wu2, wsU2, H);

    // 1) zero the aggregation buffer (d_out doubles as aggr)
    long long n4 = (long long)N * H / 4;
    egnn_zero_f4<<<(unsigned)((n4 + 255) / 256), 256, 0, stream>>>((float4*)out, n4);

    // 2) edge messages + gated scatter-add into d_out
    size_t smemE = (size_t)(K1 * LDW + H * LDW) * 2 + (size_t)WAVES * 16 * LDW * 4;
    int nEB = (E + WAVES * 16 - 1) / (WAVES * 16);
    egnn_edge_kernel<<<nEB, BLK, smemE, stream>>>(x, adj, inv, wsW1, wsW2,
                                                  bm1, bm2, We, be, out, E);

    // 3) node update + residual (reads aggr from d_out, overwrites with result)
    size_t smemU = (size_t)(2 * H * LDW + H * LDW) * 2 + (size_t)WAVES * 16 * LDW * 2;
    int nUB = (N + WAVES * 16 - 1) / (WAVES * 16);
    egnn_update_kernel<<<nUB, BLK, smemU, stream>>>(x, wsU1, wsU2, bu1, bu2, out, N);
}
